// HammingLoss_52166672777732
// MI455X (gfx1250) — 8-bit matrix operand order repaired and checked by reading, compiled, not run on hardware
//
#include <hip/hip_runtime.h>
#include <math.h>

typedef __attribute__((ext_vector_type(8))) int v8i;

#define STATIC_RATIO  1.0f
#define DYNAMIC_RATIO 1.0f
#define THRESHOLD     36.0f
#define RATIO_THR     1.1f
#define NPTS  1024
#define SBYT  32
#define HWDIM 256

__global__ __launch_bounds__(256) void k_sem1(const float* __restrict__ p,
                                              const float* __restrict__ l,
                                              float* __restrict__ part, int n) {
  __shared__ float s[256];
  const int t = threadIdx.x;
  const int base = blockIdx.x * 1024;
  float acc = 0.0f;
#pragma unroll
  for (int k = 0; k < 4; ++k) {
    int i = base + k * 256 + t;
    if (i < n) {
      float pi = p[i], li = l[i];
      float w = (li == 0.0f) ? DYNAMIC_RATIO : STATIC_RATIO;
      acc += w * (li * logf(pi) + (1.0f - li) * logf(1.0f - pi));
    }
  }
  s[t] = acc; __syncthreads();
  for (int off = 128; off > 0; off >>= 1) {
    if (t < off) s[t] += s[t + off];
    __syncthreads();
  }
  if (t == 0) part[blockIdx.x] = s[0];
}

__global__ __launch_bounds__(256) void k_sem2(const float* __restrict__ part,
                                              float* __restrict__ semOut,
                                              int nPart, float invN) {
  __shared__ float s[256];
  const int t = threadIdx.x;
  float acc = 0.0f;
  for (int i = t; i < nPart; i += 256) acc += part[i];
  s[t] = acc; __syncthreads();
  for (int off = 128; off > 0; off >>= 1) {
    if (t < off) s[t] += s[t + off];
    __syncthreads();
  }
  if (t == 0) semOut[0] = -s[0] * invN;
}

__global__ __launch_bounds__(256) void k_gather(const float* __restrict__ pred,
                                                const int* __restrict__ idxs,
                                                float* __restrict__ psel) {
  int i = blockIdx.x * 256 + threadIdx.x;
  if (i < 6 * NPTS) {
    int b3 = i >> 10;
    psel[i] = pred[b3 * (HWDIM * HWDIM) + idxs[i]];
  }
}

__global__ __launch_bounds__(32) void k_hamming(const int* __restrict__ features,
                                                int* __restrict__ best0,
                                                int* __restrict__ best1,
                                                int* __restrict__ bidx) {
  __shared__ unsigned char Abits[16 * 256];
  __shared__ unsigned char Bbits[16 * 256];
  __shared__ int Dtile[16 * 16];
  __shared__ int popQ[16], popO[16], popPart[32];

  const int tid   = threadIdx.x;
  const int rt    = blockIdx.x;
  const int combo = blockIdx.y;
  const int b     = combo >> 1;
  const int isNeg = combo & 1;
  const int qb    = (isNeg ? 4 : 2) + b;
  const int ob    = b;

  {
    const int row = tid >> 1;
    const int s0  = (tid & 1) * 16;
    const int gRow = rt * 16 + row;
    int pop = 0;
    for (int s = s0; s < s0 + 16; ++s) {
      int val = features[(qb * SBYT + s) * NPTS + gRow] & 0xFF;
      pop += __popc(val);
      unsigned int w0 = ((val >> 0) & 1) | (((val >> 1) & 1) << 8) |
                        (((val >> 2) & 1) << 16) | (((val >> 3) & 1) << 24);
      unsigned int w1 = ((val >> 4) & 1) | (((val >> 5) & 1) << 8) |
                        (((val >> 6) & 1) << 16) | (((val >> 7) & 1) << 24);
      *(unsigned int*)&Abits[row * 256 + s * 8]     = w0;
      *(unsigned int*)&Abits[row * 256 + s * 8 + 4] = w1;
    }
    popPart[tid] = pop;
  }
  __syncthreads();
  if (tid < 16) popQ[tid] = popPart[2 * tid] + popPart[2 * tid + 1];
  __syncthreads();

  int bv0 = 0x7FFFFFFF, bv1 = 0x7FFFFFFF, bi = 0;
  const int half   = tid >> 4;
  const int lane16 = tid & 15;

  for (int ct = 0; ct < 64; ++ct) {
    {
      const int col = tid >> 1;
      const int s0  = (tid & 1) * 16;
      const int gCol = ct * 16 + col;
      int pop = 0;
      for (int s = s0; s < s0 + 16; ++s) {
        int val = features[(ob * SBYT + s) * NPTS + gCol] & 0xFF;
        pop += __popc(val);
        unsigned int w0 = ((val >> 0) & 1) | (((val >> 1) & 1) << 8) |
                          (((val >> 2) & 1) << 16) | (((val >> 3) & 1) << 24);
        unsigned int w1 = ((val >> 4) & 1) | (((val >> 5) & 1) << 8) |
                          (((val >> 6) & 1) << 16) | (((val >> 7) & 1) << 24);
        *(unsigned int*)&Bbits[col * 256 + s * 8]     = w0;
        *(unsigned int*)&Bbits[col * 256 + s * 8 + 4] = w1;
      }
      popPart[tid] = pop;
    }
    __syncthreads();
    if (tid < 16) popO[tid] = popPart[2 * tid] + popPart[2 * tid + 1];
    __syncthreads();

    v8i acc = {0, 0, 0, 0, 0, 0, 0, 0};
#pragma unroll
    for (int q = 0; q < 4; ++q) {
      v8i a, bf;
#pragma unroll
      for (int g = 0; g < 8; ++g) {
        int ka = ((g >> 1) * 16) + ((g & 1) * 4) + half * 8 + q * 64;
        a[g] = *(const int*)&Abits[lane16 * 256 + ka];
        int kb = ((g >> 1) * 16) + ((g & 1) * 4) + half * 8 + q * 64;
        bf[g] = *(const int*)&Bbits[lane16 * 256 + kb];
      }
      acc = __builtin_amdgcn_wmma_i32_16x16x64_iu8(false, a, false, bf, acc,
                                                   false, false);
    }
#pragma unroll
    for (int g = 0; g < 8; ++g) {
      int row = g + 8 * half;
      int D = popQ[row] + popO[lane16] - 2 * acc[g];
      Dtile[row * 16 + lane16] = D;
    }
    __syncthreads();
    if (tid < 16) {
      for (int n = 0; n < 16; ++n) {
        int v = Dtile[tid * 16 + n];
        int j = ct * 16 + n;
        if (v < bv0)      { bv1 = bv0; bv0 = v; bi = j; }
        else if (v < bv1) { bv1 = v; }
      }
    }
    __syncthreads();
  }

  if (tid < 16) {
    int o = combo * NPTS + rt * 16 + tid;
    best0[o] = bv0; best1[o] = bv1; bidx[o] = bi;
  }
}

__device__ __forceinline__ float blockReduce(float v, float* s) {
  const int t = threadIdx.x;
  s[t] = v; __syncthreads();
  for (int off = 128; off > 0; off >>= 1) {
    if (t < off) s[t] += s[t + off];
    __syncthreads();
  }
  float r = s[0];
  __syncthreads();
  return r;
}

__global__ __launch_bounds__(256) void k_triplet(const int* __restrict__ indices,
                                                 const float* __restrict__ psel,
                                                 const int* __restrict__ best0,
                                                 const int* __restrict__ best1,
                                                 const int* __restrict__ bidx,
                                                 float* __restrict__ dpOut,
                                                 float* __restrict__ dnOut) {
  __shared__ float sred[256];
  __shared__ float Hsh[9];
  const int tid = threadIdx.x;
  const int b   = blockIdx.x;
  const int cp  = 2 * b, cn = 2 * b + 1;

  float sums[10];
#pragma unroll
  for (int q = 0; q < 10; ++q) sums[q] = 0.0f;
  for (int i = tid; i < NPTS; i += 256) {
    float b0p = (float)best0[cp * NPTS + i], b1p = (float)best1[cp * NPTS + i];
    float mp = (b0p < RATIO_THR * b1p) ? 1.0f : 0.0f;
    float b0n = (float)best0[cn * NPTS + i], b1n = (float)best1[cn * NPTS + i];
    float mn = (b0n < RATIO_THR * b1n) ? 1.0f : 0.0f;
    int loc = indices[b * NPTS + i];
    float xo = (float)(loc >> 8), yo = (float)(loc & 255);
    int ip = bidx[cp * NPTS + i];
    int lp = indices[2 * NPTS + ip];
    float up = (float)(lp >> 8), vp = (float)(lp & 255);
    int in_ = bidx[cn * NPTS + i];
    int ln = indices[4 * NPTS + in_];
    float un = (float)(ln >> 8), vn = (float)(ln & 255);
    sums[0] += mp; sums[1] += mp * xo; sums[2] += mp * yo;
    sums[3] += mp * up; sums[4] += mp * vp;
    sums[5] += mn; sums[6] += mn * xo; sums[7] += mn * yo;
    sums[8] += mn * un; sums[9] += mn * vn;
  }
  float tot[10];
  for (int q = 0; q < 10; ++q) tot[q] = blockReduce(sums[q], sred);
  float cntp = tot[0], cntn = tot[5];
  float mxo_p = tot[1] / cntp, myo_p = tot[2] / cntp;
  float mu_p  = tot[3] / cntp, mv_p  = tot[4] / cntp;
  float mxo_n = tot[6] / cntn, myo_n = tot[7] / cntn;
  float mu_n  = tot[8] / cntn, mv_n  = tot[9] / cntn;

  float mtri[36];
  float cv[8];
#pragma unroll
  for (int q = 0; q < 36; ++q) mtri[q] = 0.0f;
#pragma unroll
  for (int q = 0; q < 8; ++q) cv[q] = 0.0f;
  for (int i = tid; i < NPTS; i += 256) {
    float b0p = (float)best0[cp * NPTS + i], b1p = (float)best1[cp * NPTS + i];
    float mp = (b0p < RATIO_THR * b1p) ? 1.0f : 0.0f;
    int loc = indices[b * NPTS + i];
    float xoR = (float)(loc >> 8), yoR = (float)(loc & 255);
    int ip = bidx[cp * NPTS + i];
    int lp = indices[2 * NPTS + ip];
    float upR = (float)(lp >> 8), vpR = (float)(lp & 255);
    float xo = (xoR - mxo_p) * mp, yo = (yoR - myo_p) * mp;
    float u  = (upR - mu_p) * mp,  v  = (vpR - mv_p) * mp;
    float A0[8] = {xo, yo, mp, 0.0f, 0.0f, 0.0f, -xo * u, -yo * u};
    float A1[8] = {0.0f, 0.0f, 0.0f, xo, yo, mp, -xo * v, -yo * v};
    int t2 = 0;
#pragma unroll
    for (int i2 = 0; i2 < 8; ++i2) {
#pragma unroll
      for (int j2 = i2; j2 < 8; ++j2)
        mtri[t2++] += A0[i2] * A0[j2] + A1[i2] * A1[j2];
      cv[i2] += A0[i2] * u + A1[i2] * v;
    }
  }
  float Mf[8][9];
  {
    int t2 = 0;
    for (int i2 = 0; i2 < 8; ++i2)
      for (int j2 = i2; j2 < 8; ++j2) {
        float s = blockReduce(mtri[t2++], sred);
        if (tid == 0) { Mf[i2][j2] = s; Mf[j2][i2] = s; }
      }
    for (int i2 = 0; i2 < 8; ++i2) {
      float s = blockReduce(cv[i2], sred);
      if (tid == 0) Mf[i2][8] = s;
    }
  }
  if (tid == 0) {
    for (int k = 0; k < 8; ++k) {
      int piv = k; float mx = fabsf(Mf[k][k]);
      for (int r = k + 1; r < 8; ++r) {
        float a = fabsf(Mf[r][k]);
        if (a > mx) { mx = a; piv = r; }
      }
      if (piv != k)
        for (int c2 = k; c2 < 9; ++c2) {
          float tp = Mf[k][c2]; Mf[k][c2] = Mf[piv][c2]; Mf[piv][c2] = tp;
        }
      float inv = 1.0f / Mf[k][k];
      for (int c2 = k; c2 < 9; ++c2) Mf[k][c2] *= inv;
      for (int r = 0; r < 8; ++r) {
        if (r == k) continue;
        float f = Mf[r][k];
        for (int c2 = k; c2 < 9; ++c2) Mf[r][c2] -= f * Mf[k][c2];
      }
    }
    for (int i2 = 0; i2 < 8; ++i2) Hsh[i2] = Mf[i2][8];
    Hsh[8] = 1.0f;
  }
  __syncthreads();
  float h0 = Hsh[0], h1 = Hsh[1], h2 = Hsh[2];
  float h3 = Hsh[3], h4 = Hsh[4], h5 = Hsh[5];
  float h6 = Hsh[6], h7 = Hsh[7];

  float sp = 0.0f, sn = 0.0f;
  for (int i = tid; i < NPTS; i += 256) {
    float b0p = (float)best0[cp * NPTS + i], b1p = (float)best1[cp * NPTS + i];
    float mp = (b0p < RATIO_THR * b1p) ? 1.0f : 0.0f;
    float b0n = (float)best0[cn * NPTS + i], b1n = (float)best1[cn * NPTS + i];
    float mn = (b0n < RATIO_THR * b1n) ? 1.0f : 0.0f;
    int loc = indices[b * NPTS + i];
    float xoR = (float)(loc >> 8), yoR = (float)(loc & 255);
    int ip = bidx[cp * NPTS + i];
    int lp = indices[2 * NPTS + ip];
    float upR = (float)(lp >> 8), vpR = (float)(lp & 255);
    int in_ = bidx[cn * NPTS + i];
    int ln = indices[4 * NPTS + in_];
    float unR = (float)(ln >> 8), vnR = (float)(ln & 255);
    float pso = psel[b * NPTS + i];
    float psp = psel[(2 + b) * NPTS + ip];
    float psn = psel[(2 + b) * NPTS + in_];
    {
      float xo = (xoR - mxo_p) * mp, yo = (yoR - myo_p) * mp;
      float u  = (upR - mu_p) * mp,  v  = (vpR - mv_p) * mp;
      float sx = h0 * xo + h1 * yo + h2;
      float sy = h3 * xo + h4 * yo + h5;
      float sz = h6 * xo + h7 * yo + 1.0f;
      float px = sx / sz, py = sy / sz;
      float du = u - px, dv = v - py;
      float d = sqrtf(du * du + dv * dv + 1e-12f);
      sp += d * pso * psp * mp;
    }
    {
      float xo = (xoR - mxo_n) * mn, yo = (yoR - myo_n) * mn;
      float u  = (unR - mu_n) * mn,  v  = (vnR - mv_n) * mn;
      float sx = h0 * xo + h1 * yo + h2;
      float sy = h3 * xo + h4 * yo + h5;
      float sz = h6 * xo + h7 * yo + 1.0f;
      float px = sx / sz, py = sy / sz;
      float du = u - px, dv = v - py;
      float d = sqrtf(du * du + dv * dv + 1e-12f);
      sn += d * pso * psn * mn;
    }
  }
  float Sp = blockReduce(sp, sred);
  float Sn = blockReduce(sn, sred);
  if (tid == 0) { dpOut[b] = Sp / cntp; dnOut[b] = Sn / cntn; }
}

__global__ void k_final(const float* __restrict__ sem,
                        const float* __restrict__ dp,
                        const float* __restrict__ dn,
                        float* __restrict__ out) {
  if (threadIdx.x == 0 && blockIdx.x == 0) {
    float s = sem[0];
    float dpos = 0.5f * (dp[0] + dp[1]);
    float dneg = 0.5f * (dn[0] + dn[1]);
    float trip = fmaxf(dpos - dneg + THRESHOLD, 0.0f);
    out[0] = s + trip;
    out[1] = s;
    out[2] = trip;
  }
}

extern "C" void kernel_launch(void* const* d_in, const int* in_sizes, int n_in,
                              void* d_out, int out_size, void* d_ws, size_t ws_size,
                              hipStream_t stream) {
  const float* predictions = (const float*)d_in[0];
  const float* labels      = (const float*)d_in[1];
  const int*   indices     = (const int*)d_in[2];
  const int*   features    = (const int*)d_in[3];
  float* out = (float*)d_out;
  float* ws  = (float*)d_ws;

  const int n = in_sizes[0];
  const int nBlocks = (n + 1023) / 1024;

  float* semPart = ws;
  float* semRes  = ws + 512;
  float* psel    = ws + 640;
  int*   best0   = (int*)(ws + 7040);
  int*   best1   = (int*)(ws + 11200);
  int*   bidx    = (int*)(ws + 15360);
  float* dpOut   = ws + 19520;
  float* dnOut   = ws + 19524;

  k_sem1<<<nBlocks, 256, 0, stream>>>(predictions, labels, semPart, n);
  k_sem2<<<1, 256, 0, stream>>>(semPart, semRes, nBlocks, 1.0f / (float)n);
  k_gather<<<24, 256, 0, stream>>>(predictions, indices, psel);
  k_hamming<<<dim3(64, 4), 32, 0, stream>>>(features, best0, best1, bidx);
  k_triplet<<<2, 256, 0, stream>>>(indices, psel, best0, best1, bidx, dpOut, dnOut);
  k_final<<<1, 1, 0, stream>>>(semRes, dpOut, dnOut, out);
}
